// CrossBottleneckAttn_24988119728322
// MI455X (gfx1250) — matrix operand order repaired and checked by reading, not compiled, not run on hardware
//
#include <hip/hip_runtime.h>
#include <hip/hip_bf16.h>

typedef __attribute__((ext_vector_type(16))) __bf16 v16bf;
typedef __attribute__((ext_vector_type(8)))  __bf16 v8bf;
typedef __attribute__((ext_vector_type(8)))  float  v8f;

#define DEVINL __device__ __forceinline__

#define BB 8
#define NHEAD 4
#define DH 64
#define SEQ 1024
#define NBH 32          /* B * NHEAD */
#define HIDDEN 1024
#define DIM_OUT 256
#define ATT_SCALE 0.125f
#define LN_EPS 1e-5f

DEVINL __bf16 f2bf(float x) {
  unsigned u = __builtin_bit_cast(unsigned, x);
  unsigned r = u + 0x7fffu + ((u >> 16) & 1u);
  unsigned short h = (unsigned short)(r >> 16);
  return __builtin_bit_cast(__bf16, h);
}

DEVINL v16bf frag_from(const __bf16* lo, const __bf16* hi) {
  v8bf a = *reinterpret_cast<const v8bf*>(lo);
  v8bf b = *reinterpret_cast<const v8bf*>(hi);
  v16bf r;
#pragma unroll
  for (int i = 0; i < 8; ++i) { r[i] = a[i]; r[i + 8] = b[i]; }
  return r;
}
// A fragment (16 m x 32 k): row_k0 = &A[row m for this lane][k0]; k-contiguous rows.
DEVINL v16bf a_frag(const __bf16* row_k0, int half) {
  return frag_from(row_k0 + 8 * half, row_k0 + 16 + 8 * half);
}
// B fragment (32 k x 16 n): row_k0 = &Bt[col n for this lane][k0]; k-contiguous rows.
DEVINL v16bf b_frag(const __bf16* row_k0, int half) {
  const __bf16* p = row_k0 + 8 * half;
  return frag_from(p, p + 16);
}
DEVINL v8f wmma_bf16(v16bf a, v16bf b, v8f c) {
  return __builtin_amdgcn_wmma_f32_16x16x32_bf16(false, a, false, b, (short)0, c,
                                                 false, false);
}

// ---------------- prep kernels -------------------------------------------

__global__ void k_f32_to_bf16(const float* __restrict__ src,
                              __bf16* __restrict__ dst, int n) {
  int i = blockIdx.x * blockDim.x + threadIdx.x;
  int stride = gridDim.x * blockDim.x;
  for (; i < n; i += stride) dst[i] = f2bf(src[i]);
}

__global__ void k_rel_to_bf16(const float* __restrict__ src,
                              __bf16* __restrict__ dst) {
  int i = blockIdx.x * blockDim.x + threadIdx.x;  // 64*64
  if (i >= 64 * 64) return;
  int u = i >> 6, d = i & 63;
  dst[i] = f2bf(u < 63 ? src[u * 64 + d] : 0.0f);
}

// x: [B][512][1024] f32  ->  xT: [2][B][1024][256] bf16 (rows s, c contiguous)
__global__ void k_transpose_x(const float* __restrict__ x,
                              __bf16* __restrict__ xT) {
  long i = (long)blockIdx.x * blockDim.x + threadIdx.x;
  long n = (long)BB * 512 * SEQ;
  long stride = (long)gridDim.x * blockDim.x;
  for (; i < n; i += stride) {
    int s = (int)(i & 1023);
    long t = i >> 10;
    int c = (int)(t & 511);
    int b = (int)(t >> 9);
    int proj = c >> 8, cl = c & 255;
    xT[(((long)proj * BB + b) * SEQ + s) * 256 + cl] = f2bf(x[i]);
  }
}

// ---------------- stage 1: QKV projection (software-pipelined) ------------
// out[o,s] = sum_c W[o,c] * xT[s,c];  o<256 -> q, <512 -> kT, else vT
__global__ void k_qkv(const __bf16* __restrict__ wf, const __bf16* __restrict__ wp,
                      const __bf16* __restrict__ xT, __bf16* __restrict__ q,
                      __bf16* __restrict__ kT, __bf16* __restrict__ vT) {
  int id = blockIdx.x;
  int stile = id & 15; id >>= 4;
  int otile = id % 12; id /= 12;
  int b = id & 7;
  int proj = id >> 3;
  const __bf16* Wm = proj ? wp : wf;                         // [768][256]
  const __bf16* X = xT + ((long)proj * BB + b) * SEQ * 256;  // rows s, 256 c

  int tid = threadIdx.x, wave = tid >> 5, lane = tid & 31;
  int half = lane >> 4, nl = lane & 15;
  int o_row = otile * 64 + wave * 16 + nl;
  int s0 = stile * 64;

  const __bf16* arow = Wm + (long)o_row * 256;
  const __bf16* xrow[4];
#pragma unroll
  for (int j = 0; j < 4; ++j) xrow[j] = X + (long)(s0 + j * 16 + nl) * 256;

  v8f acc[4] = {};
  v16bf a_cur = a_frag(arow, half);
  v16bf b_cur[4];
#pragma unroll
  for (int j = 0; j < 4; ++j) b_cur[j] = b_frag(xrow[j], half);

#pragma unroll
  for (int c0 = 0; c0 < 256; c0 += 32) {
    int c1 = (c0 + 32) & 255;  // wraps to 0 on last step (discarded)
    v16bf a_nxt = a_frag(arow + c1, half);
    v16bf b_nxt[4];
#pragma unroll
    for (int j = 0; j < 4; ++j) b_nxt[j] = b_frag(xrow[j] + c1, half);
#pragma unroll
    for (int j = 0; j < 4; ++j) acc[j] = wmma_bf16(a_cur, b_cur[j], acc[j]);
    a_cur = a_nxt;
#pragma unroll
    for (int j = 0; j < 4; ++j) b_cur[j] = b_nxt[j];
  }
#pragma unroll
  for (int j = 0; j < 4; ++j) {
    int s = s0 + j * 16 + nl;
#pragma unroll
    for (int r = 0; r < 8; ++r) {
      int o = otile * 64 + wave * 16 + r + 8 * half;
      __bf16 v = f2bf(acc[j][r]);
      if (o < 256) {
        int head = o >> 6, d = o & 63;
        q[(((long)proj * NBH + b * NHEAD + head) * SEQ + s) * DH + d] = v;
      } else if (o < 512) {
        int od = o - 256, head = od >> 6, d = od & 63;
        kT[(((long)proj * NBH + b * NHEAD + head) * SEQ + s) * DH + d] = v;
      } else {
        int od = o - 512, head = od >> 6, d = od & 63;
        vT[(((long)proj * NBH + b * NHEAD + head) * DH + d) * SEQ + s] = v;
      }
    }
  }
}

// ---------------- stage 2: relative-position projections -----------------
// P[proj][which][bh][n][u] = sum_d q[proj][bh][n][d] * rel[u][d]
__global__ void k_relproj(const __bf16* __restrict__ q,
                          const __bf16* __restrict__ hr,
                          const __bf16* __restrict__ wr,
                          float* __restrict__ P) {
  int id = blockIdx.x;
  int ntile = id & 15; id >>= 4;
  int bh = id & 31; id >>= 5;
  int which = id & 1;
  int proj = id >> 1;
  const __bf16* rel = which ? wr : hr;  // [64][64] (row 63 zero)
  const __bf16* Q = q + ((long)proj * NBH + bh) * SEQ * DH;
  float* Pout = P + (((long)proj * 2 + which) * NBH + bh) * (long)SEQ * 64;

  int tid = threadIdx.x, wave = tid >> 5, lane = tid & 31;
  int half = lane >> 4, nl = lane & 15;
  int n_row = ntile * 64 + wave * 16 + nl;

  const __bf16* arow = Q + (long)n_row * DH;
  v16bf a0 = a_frag(arow, half);
  v16bf a1 = a_frag(arow + 32, half);
  v16bf b0[4], b1[4];
#pragma unroll
  for (int j = 0; j < 4; ++j) {
    const __bf16* rrow = rel + (j * 16 + nl) * 64;
    b0[j] = b_frag(rrow, half);
    b1[j] = b_frag(rrow + 32, half);
  }
  v8f acc[4] = {};
#pragma unroll
  for (int j = 0; j < 4; ++j) {
    acc[j] = wmma_bf16(a0, b0[j], acc[j]);
    acc[j] = wmma_bf16(a1, b1[j], acc[j]);
  }
#pragma unroll
  for (int j = 0; j < 4; ++j) {
    int u = j * 16 + nl;
#pragma unroll
    for (int r = 0; r < 8; ++r) {
      int n = ntile * 64 + wave * 16 + r + 8 * half;
      Pout[(long)n * 64 + u] = acc[j][r];
    }
  }
}

// ---------------- stage 3: flash cross-attention (pipelined K/V) ----------
__global__ void k_attn(const __bf16* __restrict__ q, const __bf16* __restrict__ kT,
                       const __bf16* __restrict__ vT, const float* __restrict__ P,
                       float* __restrict__ xln, float* __restrict__ sums) {
  int id = blockIdx.x;
  int ntile = id & 15; id >>= 4;
  int bh = id & 31;
  int proj = id >> 5;  // 0 = out_f (q_f, k_p, v_p), 1 = out_p
  const __bf16* Q = q + ((long)proj * NBH + bh) * SEQ * DH;
  const __bf16* K = kT + ((long)(1 - proj) * NBH + bh) * SEQ * DH;
  const __bf16* V = vT + ((long)(1 - proj) * NBH + bh) * DH * SEQ;
  const float* Ph = P + (((long)proj * 2 + 0) * NBH + bh) * (long)SEQ * 64;
  const float* Pw = P + (((long)proj * 2 + 1) * NBH + bh) * (long)SEQ * 64;

  int tid = threadIdx.x, wave = tid >> 5, lane = tid & 31;
  int half = lane >> 4, nl = lane & 15;
  int n0 = ntile * 64 + wave * 16;

  __shared__ __align__(16) __bf16 Pbuf[4][16][40];

  const __bf16* qrow = Q + (long)(n0 + nl) * DH;
  v16bf a0 = a_frag(qrow, half);       // d = 0..31
  v16bf a1 = a_frag(qrow + 32, half);  // d = 32..63

  v8f o0 = {}, o1 = {}, o2 = {}, o3 = {};
  float mx[8], l[8];
  int n_of_r[8];
#pragma unroll
  for (int r = 0; r < 8; ++r) {
    mx[r] = -1e30f;
    l[r] = 0.0f;
    n_of_r[r] = n0 + r + 8 * half;
  }

  // prime K fragments for m0 = 0
  v16bf bk[4];
  {
    const __bf16* k0row = K + (long)nl * DH;
    const __bf16* k1row = K + (long)(16 + nl) * DH;
    bk[0] = b_frag(k0row, half);
    bk[1] = b_frag(k0row + 32, half);
    bk[2] = b_frag(k1row, half);
    bk[3] = b_frag(k1row + 32, half);
  }

  for (int m0 = 0; m0 < SEQ; m0 += 32) {
    // --- scores with resident K fragments ---
    v8f s0v = {}, s1v = {};
    s0v = wmma_bf16(a0, bk[0], s0v);
    s0v = wmma_bf16(a1, bk[1], s0v);
    s1v = wmma_bf16(a0, bk[2], s1v);
    s1v = wmma_bf16(a1, bk[3], s1v);

    // --- issue next-step K loads and this-step V loads; softmax hides them ---
    int mn = (m0 + 32) & (SEQ - 1);  // wraps on last step (discarded)
    const __bf16* k0row = K + (long)(mn + nl) * DH;
    const __bf16* k1row = K + (long)(mn + 16 + nl) * DH;
    v16bf nk0 = b_frag(k0row, half);
    v16bf nk1 = b_frag(k0row + 32, half);
    v16bf nk2 = b_frag(k1row, half);
    v16bf nk3 = b_frag(k1row + 32, half);
    v16bf bv[4];
#pragma unroll
    for (int j = 0; j < 4; ++j)
      bv[j] = b_frag(V + (long)(j * 16 + nl) * SEQ + m0, half);

    // --- positional logits + online softmax ---
    float e0[8], e1[8], alpha[8];
#pragma unroll
    for (int r = 0; r < 8; ++r) {
      int n = n_of_r[r];
      int hn = n >> 5, wn = n & 31;
      int ma = m0 + nl, mb = m0 + 16 + nl;
      e0[r] = s0v[r] * ATT_SCALE + Ph[(long)n * 64 + ((ma >> 5) - hn + 31)] +
              Pw[(long)n * 64 + ((ma & 31) - wn + 31)];
      e1[r] = s1v[r] * ATT_SCALE + Ph[(long)n * 64 + ((mb >> 5) - hn + 31)] +
              Pw[(long)n * 64 + ((mb & 31) - wn + 31)];
    }
#pragma unroll
    for (int r = 0; r < 8; ++r) {
      float tm = fmaxf(e0[r], e1[r]);
#pragma unroll
      for (int msk = 1; msk < 16; msk <<= 1)
        tm = fmaxf(tm, __shfl_xor(tm, msk, 32));
      float nm = fmaxf(mx[r], tm);
      alpha[r] = __expf(mx[r] - nm);
      mx[r] = nm;
      e0[r] = __expf(e0[r] - nm);
      e1[r] = __expf(e1[r] - nm);
      float rs = e0[r] + e1[r];
#pragma unroll
      for (int msk = 1; msk < 16; msk <<= 1) rs += __shfl_xor(rs, msk, 32);
      l[r] = l[r] * alpha[r] + rs;
    }
#pragma unroll
    for (int r = 0; r < 8; ++r) {
      o0[r] *= alpha[r]; o1[r] *= alpha[r];
      o2[r] *= alpha[r]; o3[r] *= alpha[r];
    }
    // transpose prob tile (C layout) -> A layout via per-wave LDS buffer
#pragma unroll
    for (int r = 0; r < 8; ++r) {
      Pbuf[wave][r + 8 * half][nl] = f2bf(e0[r]);
      Pbuf[wave][r + 8 * half][16 + nl] = f2bf(e1[r]);
    }
    asm volatile("s_wait_dscnt 0x0" ::: "memory");
    v16bf ap = a_frag(&Pbuf[wave][nl][0], half);

    o0 = wmma_bf16(ap, bv[0], o0);
    o1 = wmma_bf16(ap, bv[1], o1);
    o2 = wmma_bf16(ap, bv[2], o2);
    o3 = wmma_bf16(ap, bv[3], o3);

    bk[0] = nk0; bk[1] = nk1; bk[2] = nk2; bk[3] = nk3;
  }

  int head = bh & 3, b = bh >> 2;
  int c0 = proj * 256 + head * 64;
  float lsum = 0.0f, lsq = 0.0f;
#pragma unroll
  for (int r = 0; r < 8; ++r) {
    float inv = 1.0f / l[r];
    float v0 = o0[r] * inv, v1 = o1[r] * inv, v2 = o2[r] * inv, v3 = o3[r] * inv;
    int n = n_of_r[r];
    xln[((long)b * 512 + c0 + 0 * 16 + nl) * SEQ + n] = v0;
    xln[((long)b * 512 + c0 + 1 * 16 + nl) * SEQ + n] = v1;
    xln[((long)b * 512 + c0 + 2 * 16 + nl) * SEQ + n] = v2;
    xln[((long)b * 512 + c0 + 3 * 16 + nl) * SEQ + n] = v3;
    lsum += v0 + v1 + v2 + v3;
    lsq += v0 * v0 + v1 * v1 + v2 * v2 + v3 * v3;
  }
#pragma unroll
  for (int msk = 1; msk < 32; msk <<= 1) {
    lsum += __shfl_xor(lsum, msk, 32);
    lsq += __shfl_xor(lsq, msk, 32);
  }
  if (lane == 0) {
    atomicAdd(&sums[b * 2 + 0], lsum);
    atomicAdd(&sums[b * 2 + 1], lsq);
  }
}

// ---------------- stage 4a: LayerNorm + transpose to bf16 -----------------
__global__ void k_ln(const float* __restrict__ xln, const float* __restrict__ sums,
                     const float* __restrict__ gamma, const float* __restrict__ beta,
                     __bf16* __restrict__ xhatT) {
  long i = (long)blockIdx.x * blockDim.x + threadIdx.x;
  long n = (long)BB * 512 * SEQ;
  long stride = (long)gridDim.x * blockDim.x;
  const float invN = 1.0f / (512.0f * 1024.0f);
  for (; i < n; i += stride) {
    int s = (int)(i & 1023);
    long t = i >> 10;
    int c = (int)(t & 511);
    int b = (int)(t >> 9);
    float mean = sums[b * 2 + 0] * invN;
    float var = sums[b * 2 + 1] * invN - mean * mean;
    float rstd = rsqrtf(var + LN_EPS);
    float v = (xln[i] - mean) * rstd * gamma[(long)c * SEQ + s] +
              beta[(long)c * SEQ + s];
    xhatT[((long)b * SEQ + s) * 512 + c] = f2bf(v);
  }
}

// ---------------- stage 4: fc1 + GELU (software-pipelined) ----------------
__global__ void k_fc1(const __bf16* __restrict__ w1, const float* __restrict__ b1,
                      const __bf16* __restrict__ xhatT, __bf16* __restrict__ hT) {
  int id = blockIdx.x;
  int stile = id & 15; id >>= 4;
  int otile = id & 15;
  int b = id >> 4;
  int tid = threadIdx.x, wave = tid >> 5, lane = tid & 31;
  int half = lane >> 4, nl = lane & 15;
  int o_row = otile * 64 + wave * 16 + nl;
  int s0 = stile * 64;
  const __bf16* arow = w1 + (long)o_row * 512;
  const __bf16* X = xhatT + (long)b * SEQ * 512;
  const __bf16* xrow[4];
#pragma unroll
  for (int j = 0; j < 4; ++j) xrow[j] = X + (long)(s0 + j * 16 + nl) * 512;

  v8f acc[4] = {};
  v16bf a_cur = a_frag(arow, half);
  v16bf b_cur[4];
#pragma unroll
  for (int j = 0; j < 4; ++j) b_cur[j] = b_frag(xrow[j], half);

  for (int c0 = 0; c0 < 512; c0 += 32) {
    int c1 = (c0 + 32) & 511;
    v16bf a_nxt = a_frag(arow + c1, half);
    v16bf b_nxt[4];
#pragma unroll
    for (int j = 0; j < 4; ++j) b_nxt[j] = b_frag(xrow[j] + c1, half);
#pragma unroll
    for (int j = 0; j < 4; ++j) acc[j] = wmma_bf16(a_cur, b_cur[j], acc[j]);
    a_cur = a_nxt;
#pragma unroll
    for (int j = 0; j < 4; ++j) b_cur[j] = b_nxt[j];
  }
#pragma unroll
  for (int j = 0; j < 4; ++j) {
    int s = s0 + j * 16 + nl;
#pragma unroll
    for (int r = 0; r < 8; ++r) {
      int o = otile * 64 + wave * 16 + r + 8 * half;
      float v = acc[j][r] + b1[o];
      v = 0.5f * v * (1.0f + erff(v * 0.70710678118f));  // exact GELU
      hT[((long)b * SEQ + s) * HIDDEN + o] = f2bf(v);
    }
  }
}

// ---------------- stage 5: fc2 (software-pipelined) -----------------------
__global__ void k_fc2(const __bf16* __restrict__ w2, const float* __restrict__ b2,
                      const __bf16* __restrict__ hT, float* __restrict__ out) {
  int id = blockIdx.x;
  int stile = id & 15; id >>= 4;
  int otile = id & 3;
  int b = id >> 2;
  int tid = threadIdx.x, wave = tid >> 5, lane = tid & 31;
  int half = lane >> 4, nl = lane & 15;
  int o_row = otile * 64 + wave * 16 + nl;
  int s0 = stile * 64;
  const __bf16* arow = w2 + (long)o_row * HIDDEN;
  const __bf16* X = hT + (long)b * SEQ * HIDDEN;
  const __bf16* xrow[4];
#pragma unroll
  for (int j = 0; j < 4; ++j) xrow[j] = X + (long)(s0 + j * 16 + nl) * HIDDEN;

  v8f acc[4] = {};
  v16bf a_cur = a_frag(arow, half);
  v16bf b_cur[4];
#pragma unroll
  for (int j = 0; j < 4; ++j) b_cur[j] = b_frag(xrow[j], half);

  for (int c0 = 0; c0 < HIDDEN; c0 += 32) {
    int c1 = (c0 + 32) & (HIDDEN - 1);
    v16bf a_nxt = a_frag(arow + c1, half);
    v16bf b_nxt[4];
#pragma unroll
    for (int j = 0; j < 4; ++j) b_nxt[j] = b_frag(xrow[j] + c1, half);
#pragma unroll
    for (int j = 0; j < 4; ++j) acc[j] = wmma_bf16(a_cur, b_cur[j], acc[j]);
    a_cur = a_nxt;
#pragma unroll
    for (int j = 0; j < 4; ++j) b_cur[j] = b_nxt[j];
  }
#pragma unroll
  for (int j = 0; j < 4; ++j) {
    int s = s0 + j * 16 + nl;
#pragma unroll
    for (int r = 0; r < 8; ++r) {
      int o = otile * 64 + wave * 16 + r + 8 * half;
      out[((long)b * DIM_OUT + o) * SEQ + s] = acc[j][r] + b2[o];
    }
  }
}

// ---------------- host launcher ------------------------------------------
extern "C" void kernel_launch(void* const* d_in, const int* in_sizes, int n_in,
                              void* d_out, int out_size, void* d_ws, size_t ws_size,
                              hipStream_t stream) {
  const float* x = (const float*)d_in[0];
  const float* qkv_f_w = (const float*)d_in[1];
  const float* qkv_p_w = (const float*)d_in[2];
  const float* height_rel = (const float*)d_in[3];
  const float* width_rel = (const float*)d_in[4];
  const float* ln_gamma = (const float*)d_in[5];
  const float* ln_beta = (const float*)d_in[6];
  const float* fc1_w = (const float*)d_in[7];
  const float* fc1_b = (const float*)d_in[8];
  const float* fc2_w = (const float*)d_in[9];
  const float* fc2_b = (const float*)d_in[10];

  char* ws = (char*)d_ws;
  size_t o_wf = 0;
  size_t o_wp = o_wf + 768 * 256 * 2;
  size_t o_w1 = o_wp + 768 * 256 * 2;
  size_t o_w2 = o_w1 + (size_t)1024 * 512 * 2;
  size_t o_hr = o_w2 + (size_t)256 * 1024 * 2;
  size_t o_wr = o_hr + 64 * 64 * 2;
  size_t o_sums = o_wr + 64 * 64 * 2;
  size_t o_xT = o_sums + 256;                          // also reused as xhatT
  size_t o_q = o_xT + (size_t)2 * BB * SEQ * 256 * 2;
  size_t o_kT = o_q + (size_t)2 * NBH * SEQ * DH * 2;
  size_t o_vT = o_kT + (size_t)2 * NBH * SEQ * DH * 2;
  size_t o_P = o_vT + (size_t)2 * NBH * DH * SEQ * 2;  // also reused as hT
  size_t o_xln = o_P + (size_t)4 * NBH * SEQ * 64 * 4;

  __bf16* wf_bf = (__bf16*)(ws + o_wf);
  __bf16* wp_bf = (__bf16*)(ws + o_wp);
  __bf16* w1_bf = (__bf16*)(ws + o_w1);
  __bf16* w2_bf = (__bf16*)(ws + o_w2);
  __bf16* hr_bf = (__bf16*)(ws + o_hr);
  __bf16* wr_bf = (__bf16*)(ws + o_wr);
  float* sums = (float*)(ws + o_sums);
  __bf16* xT = (__bf16*)(ws + o_xT);
  __bf16* qb = (__bf16*)(ws + o_q);
  __bf16* kTb = (__bf16*)(ws + o_kT);
  __bf16* vTb = (__bf16*)(ws + o_vT);
  float* Pbuf = (float*)(ws + o_P);
  float* xln = (float*)(ws + o_xln);
  __bf16* xhatT = (__bf16*)(ws + o_xT);  // overlays xT (same size, xT dead)
  __bf16* hT = (__bf16*)(ws + o_P);      // overlays P (P dead after attention)

  // prep: weight/rel conversion, x transpose, clear LN sums
  k_f32_to_bf16<<<(768 * 256 + 255) / 256, 256, 0, stream>>>(qkv_f_w, wf_bf, 768 * 256);
  k_f32_to_bf16<<<(768 * 256 + 255) / 256, 256, 0, stream>>>(qkv_p_w, wp_bf, 768 * 256);
  k_f32_to_bf16<<<(1024 * 512 + 255) / 256, 256, 0, stream>>>(fc1_w, w1_bf, 1024 * 512);
  k_f32_to_bf16<<<(256 * 1024 + 255) / 256, 256, 0, stream>>>(fc2_w, w2_bf, 256 * 1024);
  k_rel_to_bf16<<<16, 256, 0, stream>>>(height_rel, hr_bf);
  k_rel_to_bf16<<<16, 256, 0, stream>>>(width_rel, wr_bf);
  k_transpose_x<<<4096, 256, 0, stream>>>(x, xT);
  hipMemsetAsync(sums, 0, 16 * sizeof(float), stream);

  // stage 1: QKV projection (2 proj x 8 b x 12 otiles x 16 stiles)
  k_qkv<<<2 * 8 * 12 * 16, 128, 0, stream>>>(wf_bf, wp_bf, xT, qb, kTb, vTb);
  // stage 2: rel-position projections (2 proj x 2 which x 32 bh x 16 ntiles)
  k_relproj<<<2 * 2 * 32 * 16, 128, 0, stream>>>(qb, hr_bf, wr_bf, Pbuf);
  // stage 3: flash cross-attention (2 proj x 32 bh x 16 ntiles)
  k_attn<<<2 * 32 * 16, 128, 0, stream>>>(qb, kTb, vTb, Pbuf, xln, sums);
  // stage 4a: LayerNorm
  k_ln<<<4096, 256, 0, stream>>>(xln, sums, ln_gamma, ln_beta, xhatT);
  // stage 4: fc1 + GELU (8 b x 16 otiles x 16 stiles)
  k_fc1<<<8 * 16 * 16, 128, 0, stream>>>(w1_bf, fc1_b, xhatT, hT);
  // stage 5: fc2 (8 b x 4 otiles x 16 stiles)
  k_fc2<<<8 * 4 * 16, 128, 0, stream>>>(w2_bf, fc2_b, hT, (float*)d_out);
}
